// LSTM_model_36773509988383
// MI455X (gfx1250) — matrix operand order repaired and checked by reading, not compiled, not run on hardware
//
#include <hip/hip_runtime.h>
#include <hip/hip_bf16.h>

// ---------------------------------------------------------------------------
// LSTM for MI455X (gfx1250, wave32, WMMA).
//   1) convert x -> bf16; pack W,U,V into WMMA-B fragment-major bf16 tiles
//   2) xW = x @ W + hid_bias   (big bf16 WMMA GEMM, f32 out in workspace)
//   3) 512 sequential step kernels: gates = xW[:,t,:] + h@U. Each step is
//      spread over 64 WGPs (64 blocks x 8 waves); each wave takes a K-slice
//      of 128 (4 WMMA k-iters, 8 short independent accumulator chains),
//      partials reduced in LDS by all 8 waves in parallel, then a fully
//      data-parallel fused i/f/g/o -> c/h epilogue across all 256 threads.
//   4) output = hs @ V + out_bias (bf16 WMMA GEMM)
// ---------------------------------------------------------------------------

typedef __bf16 bf16;
typedef __bf16 v16bf __attribute__((ext_vector_type(16)));
typedef __bf16 v8bf  __attribute__((ext_vector_type(8)));
typedef float  v8f   __attribute__((ext_vector_type(8)));

union FragAB { v16bf v; v8bf h[2]; };

__device__ __forceinline__ float fast_sigmoid(float x) {
    return 1.0f / (1.0f + __expf(-x));
}
__device__ __forceinline__ float fast_tanh(float x) {
    x = fminf(10.0f, fmaxf(-10.0f, x));
    float e = __expf(2.0f * x);
    return (e - 1.0f) / (e + 1.0f);
}

__device__ __forceinline__ v8f wmma_bf16(v16bf a, v16bf b, v8f c) {
    // v_wmma_f32_16x16x32_bf16  (neg_a, A, neg_b, B, c_mod, C, reuse_a, reuse_b)
    return __builtin_amdgcn_wmma_f32_16x16x32_bf16(false, a, false, b, (short)0, c,
                                                   false, false);
}

// A fragment, row-major bf16 [M x K] (lda = K). Contiguous k order used here (same as packed B):
// lane<16: K = k0 + {0..15}; lane>=16: K = k0 + {16..31}; row = lane%16.
__device__ __forceinline__ v16bf load_a_frag(const bf16* __restrict__ A, int lda,
                                             int m0, int k0, int lane) {
    const int half = lane >> 4;
    const int row  = m0 + (lane & 15);
    const bf16* p  = A + (long)row * lda + k0 + half * 16;
    FragAB f;
    f.h[0] = *(const v8bf*)(p);       // K = k0 + 16*half + 0..7
    f.h[1] = *(const v8bf*)(p + 8);   // K = k0 + 16*half + 8..15
    return f.v;
}

// B fragment from packed tiles: tile (kt,nt) stored as 32 lanes x 16 bf16,
// lane l holds B[kt*32 + 16*(l>>4) + j][nt*16 + (l&15)], j = 0..15.
__device__ __forceinline__ v16bf load_b_frag(const bf16* __restrict__ Bp,
                                             long tileIdx, int lane) {
    const bf16* p = Bp + tileIdx * 512 + lane * 16;
    FragAB f;
    f.h[0] = *(const v8bf*)(p);
    f.h[1] = *(const v8bf*)(p + 8);
    return f.v;
}

// ----------------------------- prep kernels --------------------------------

__global__ void k_f32_to_bf16(const float* __restrict__ src, bf16* __restrict__ dst, int n) {
    int i = blockIdx.x * 256 + threadIdx.x;
    if (i < n) dst[i] = (bf16)src[i];
}

// Pack f32 row-major [K x N] into WMMA-B fragment-major bf16 tiles.
__global__ void k_pack_b(const float* __restrict__ src, bf16* __restrict__ dst,
                         int K, int N) {
    int idx   = blockIdx.x * 256 + threadIdx.x;
    int total = (K >> 5) * (N >> 4) * 512;
    if (idx >= total) return;
    int tile = idx >> 9;
    int r    = idx & 511;
    int lane = r >> 4;
    int j    = r & 15;
    int ntn  = N >> 4;
    int kt   = tile / ntn;
    int nt   = tile - kt * ntn;
    int k    = kt * 32 + ((lane >> 4) << 4) + j;
    int n    = nt * 16 + (lane & 15);
    dst[idx] = (bf16)src[(long)k * N + n];
}

__global__ void k_init_state(bf16* __restrict__ h0, float* __restrict__ c0, int n) {
    int i = blockIdx.x * 256 + threadIdx.x;
    if (i < n) { h0[i] = (bf16)0.0f; c0[i] = 0.0f; }
}

// ----------------------------- GEMM kernel ---------------------------------
// C[M x N] = A[M x K](bf16, row-major) @ Bp(packed bf16) + bias[N]
// 256 threads = 8 waves (2 along M x 4 along N); wave tile 32x32 (4 frags);
// block tile 64x128.
__global__ void k_gemm_bf16(const bf16* __restrict__ A, const bf16* __restrict__ Bp,
                            const float* __restrict__ bias, float* __restrict__ C,
                            int M, int N, int K) {
    const int lane = threadIdx.x & 31;
    const int wave = threadIdx.x >> 5;
    const int wn   = wave & 3;
    const int wm   = wave >> 2;
    const int m0   = blockIdx.y * 64 + wm * 32;
    const int n0   = blockIdx.x * 128 + wn * 32;
    const int ntn  = N >> 4;

    v8f acc00 = {}, acc01 = {}, acc10 = {}, acc11 = {};

    for (int k0 = 0; k0 < K; k0 += 32) {
        const int kt = k0 >> 5;
        v16bf a0 = load_a_frag(A, K, m0,      k0, lane);
        v16bf a1 = load_a_frag(A, K, m0 + 16, k0, lane);
        v16bf b0 = load_b_frag(Bp, (long)kt * ntn + (n0 >> 4),     lane);
        v16bf b1 = load_b_frag(Bp, (long)kt * ntn + (n0 >> 4) + 1, lane);
        acc00 = wmma_bf16(a0, b0, acc00);
        acc10 = wmma_bf16(a1, b0, acc10);
        acc01 = wmma_bf16(a0, b1, acc01);
        acc11 = wmma_bf16(a1, b1, acc11);
        if (k0 + 32 < K) {  // hint next K-slab into cache (global_prefetch_b8)
            __builtin_prefetch(A + (long)(m0 + (lane & 15)) * K + k0 + 32, 0, 1);
            __builtin_prefetch(Bp + ((long)(kt + 1) * ntn + (n0 >> 4)) * 512 + lane * 16, 0, 1);
        }
    }

    const int half = lane >> 4;
    const int nc   = n0 + (lane & 15);
#pragma unroll
    for (int mt = 0; mt < 2; ++mt) {
        v8f am0 = (mt == 0) ? acc00 : acc10;
        v8f am1 = (mt == 0) ? acc01 : acc11;
#pragma unroll
        for (int r = 0; r < 8; ++r) {
            int m = m0 + mt * 16 + r + half * 8;  // C layout: VGPR r -> row r (+8 for hi half)
            C[(long)m * N + nc]      = am0[r] + bias[nc];
            C[(long)m * N + nc + 16] = am1[r] + bias[nc + 16];
        }
    }
}

// ----------------------------- LSTM step -----------------------------------
// gates[32 x 4096] = h_prev @ U (WMMA) + xW[:, t, :].
// Grid: 64 blocks (one 16-wide hidden-column tile each, ~one per WGP),
// 256 threads = 8 waves. Wave w accumulates K-slice [w*128, w*128+128)
// for all 4 gates x 2 M-tiles. Reduction: wave f sums fragment f across the
// 8 partials (parallel). Epilogue: all 256 threads handle 2 cells each.
__global__ void k_lstm_step(const bf16* __restrict__ hprev, const bf16* __restrict__ Up,
                            const float* __restrict__ xW, int t,
                            float* __restrict__ c_state,
                            bf16* __restrict__ hnext, bf16* __restrict__ hs_bf,
                            float* __restrict__ out_hs, float* __restrict__ out_cs,
                            float* __restrict__ out_fs, float* __restrict__ out_fs2,
                            float* __restrict__ out_gs, float* __restrict__ out_os) {
    const int lane = threadIdx.x & 31;
    const int wave = threadIdx.x >> 5;      // 0..7 : K-slice owner / reduce frag
    const int j0   = blockIdx.x * 16;       // hidden-column tile (0..1008)
    const int ntn  = 256;                   // 4096/16 tiles along N of U

    // partial sums: [wave][frag = mt*4+g][lane][8]  -> 64 KB of WGP LDS
    __shared__ float red[8 * 8 * 32 * 8];

    // warm the xW slice this block consumes (overlaps the WMMA phase)
    {
        int seg = threadIdx.x;              // 128 segments: (m, gate) pairs
        if (seg < 128) {
            int m = seg >> 2, g = seg & 3;
            __builtin_prefetch(xW + ((long)m * 512 + t) * 4096 + g * 1024 + j0, 0, 1);
        }
    }

    v8f acc[2][4];
#pragma unroll
    for (int mt = 0; mt < 2; ++mt)
#pragma unroll
        for (int g = 0; g < 4; ++g) acc[mt][g] = (v8f){};

    const int kbase = wave * 128;           // this wave's K-slice
#pragma unroll
    for (int ki = 0; ki < 4; ++ki) {
        const int k0 = kbase + ki * 32;
        const int kt = k0 >> 5;
        v16bf a0 = load_a_frag(hprev, 1024, 0,  k0, lane);
        v16bf a1 = load_a_frag(hprev, 1024, 16, k0, lane);
#pragma unroll
        for (int g = 0; g < 4; ++g) {
            long tile = (long)kt * ntn + ((g * 1024 + j0) >> 4);
            v16bf b   = load_b_frag(Up, tile, lane);
            acc[0][g] = wmma_bf16(a0, b, acc[0][g]);
            acc[1][g] = wmma_bf16(a1, b, acc[1][g]);
        }
    }

    // all waves deposit their partials
#pragma unroll
    for (int mt = 0; mt < 2; ++mt)
#pragma unroll
        for (int g = 0; g < 4; ++g) {
            int frag = mt * 4 + g;
            *(v8f*)&red[((wave * 8 + frag) * 32 + lane) * 8] = acc[mt][g];
        }
    __syncthreads();

    // wave f reduces fragment f across the 8 K-slices (vectorized, parallel)
    {
        v8f s = *(const v8f*)&red[((0 * 8 + wave) * 32 + lane) * 8];
#pragma unroll
        for (int w2 = 1; w2 < 8; ++w2)
            s += *(const v8f*)&red[((w2 * 8 + wave) * 32 + lane) * 8];
        *(v8f*)&red[((0 * 8 + wave) * 32 + lane) * 8] = s;
    }
    __syncthreads();

    // data-parallel fused cell update: 512 cells (32 batch x 16 cols),
    // 2 per thread; consecutive threads -> consecutive j (coalesced).
#pragma unroll
    for (int cc = 0; cc < 2; ++cc) {
        const int cell = threadIdx.x + cc * 256;  // 0..511
        const int m    = cell >> 4;               // batch row 0..31
        const int jj   = cell & 15;
        const int mt   = m >> 4;
        const int rm   = m & 15;
        const int r    = rm & 7;
        const int lsrc = jj + ((rm >> 3) << 4);   // source lane in C layout
        const int j    = j0 + jj;

        const float gi = red[(((mt * 4 + 0)) * 32 + lsrc) * 8 + r];
        const float gf = red[(((mt * 4 + 1)) * 32 + lsrc) * 8 + r];
        const float gg = red[(((mt * 4 + 2)) * 32 + lsrc) * 8 + r];
        const float go = red[(((mt * 4 + 3)) * 32 + lsrc) * 8 + r];

        const long xrow = ((long)m * 512 + t) * 4096;
        float iv = fast_sigmoid(gi + xW[xrow + j]);
        float fv = fast_sigmoid(gf + xW[xrow + 1024 + j]);
        float gv = fast_tanh   (gg + xW[xrow + 2048 + j]);
        float ov = fast_sigmoid(go + xW[xrow + 3072 + j]);

        const long sidx = (long)m * 1024 + j;
        float cv = fv * c_state[sidx] + iv * gv;
        float hv = ov * fast_tanh(cv);
        c_state[sidx] = cv;
        hnext[sidx]   = (bf16)hv;
        const long oidx = ((long)m * 512 + t) * 1024 + j;  // [B,T,H]
        hs_bf[oidx]   = (bf16)hv;
        out_hs[oidx]  = hv;
        out_cs[oidx]  = cv;
        out_fs[oidx]  = fv;
        out_fs2[oidx] = fv;
        out_gs[oidx]  = gv;
        out_os[oidx]  = ov;
    }
}

// ----------------------------- launch --------------------------------------

extern "C" void kernel_launch(void* const* d_in, const int* in_sizes, int n_in,
                              void* d_out, int out_size, void* d_ws, size_t ws_size,
                              hipStream_t stream) {
    (void)in_sizes; (void)n_in; (void)out_size; (void)ws_size;

    const float* x        = (const float*)d_in[0];
    const float* W        = (const float*)d_in[1];
    const float* U        = (const float*)d_in[2];
    const float* hid_bias = (const float*)d_in[3];
    const float* V        = (const float*)d_in[4];
    const float* out_bias = (const float*)d_in[5];
    float* out = (float*)d_out;

    constexpr int  Bz = 32, T = 512, NI = 512, H = 1024, NO = 512;
    constexpr int  BT = Bz * T;                      // 16384
    constexpr long HS = (long)Bz * T * H;            // 16,777,216

    // d_out: (hs, output, cs, fs, fs, gs, os) concatenated flat
    float* out_hs  = out;
    float* out_y   = out_hs + HS;
    float* out_cs  = out_y + (long)Bz * T * NO;
    float* out_fs  = out_cs + HS;
    float* out_fs2 = out_fs + HS;
    float* out_gs  = out_fs2 + HS;
    float* out_os  = out_gs + HS;

    // workspace carve (256B aligned regions)
    char* wp = (char*)d_ws;
    auto carve = [&](size_t bytes) -> void* {
        void* p = wp; wp += (bytes + 255) & ~(size_t)255; return p;
    };
    float* xW     = (float*)carve((size_t)BT * 4096 * 4);  // 256 MB
    bf16*  xb     = (bf16*) carve((size_t)BT * NI * 2);
    bf16*  Wp     = (bf16*) carve((size_t)NI * 4096 * 2);
    bf16*  Up     = (bf16*) carve((size_t)H  * 4096 * 2);
    bf16*  Vp     = (bf16*) carve((size_t)H  * NO   * 2);
    bf16*  hs_bf  = (bf16*) carve((size_t)BT * H * 2);
    bf16*  hbuf0  = (bf16*) carve((size_t)Bz * H * 2);
    bf16*  hbuf1  = (bf16*) carve((size_t)Bz * H * 2);
    float* cstate = (float*)carve((size_t)Bz * H * 4);

    // 1) convert / pack / init
    k_f32_to_bf16<<<(BT * NI + 255) / 256, 256, 0, stream>>>(x, xb, BT * NI);
    k_pack_b<<<((NI / 32) * (4096 / 16) * 512 + 255) / 256, 256, 0, stream>>>(W, Wp, NI, 4096);
    k_pack_b<<<((H  / 32) * (4096 / 16) * 512 + 255) / 256, 256, 0, stream>>>(U, Up, H, 4096);
    k_pack_b<<<((H  / 32) * (NO   / 16) * 512 + 255) / 256, 256, 0, stream>>>(V, Vp, H, NO);
    k_init_state<<<(Bz * H + 255) / 256, 256, 0, stream>>>(hbuf0, cstate, Bz * H);

    // 2) xW = x @ W + hid_bias  -> [16384 x 4096] f32
    dim3 g1(4096 / 128, BT / 64);
    k_gemm_bf16<<<g1, 256, 0, stream>>>(xb, Wp, hid_bias, xW, BT, 4096, NI);

    // 3) recurrence: 512 fused cell-update kernels, each spread over 64 WGPs
    for (int t = 0; t < T; ++t) {
        const bf16* hp = (t & 1) ? hbuf1 : hbuf0;
        bf16*       hn = (t & 1) ? hbuf0 : hbuf1;
        k_lstm_step<<<64, 256, 0, stream>>>(hp, Up, xW, t, cstate, hn, hs_bf,
                                            out_hs, out_cs, out_fs, out_fs2,
                                            out_gs, out_os);
    }

    // 4) output = hs @ V + out_bias -> [16384 x 512]
    dim3 g3(NO / 128, BT / 64);
    k_gemm_bf16<<<g3, 256, 0, stream>>>(hs_bf, Vp, out_bias, out_y, BT, NO, H);
}
